// UDPointerParser_57775900066486
// MI455X (gfx1250) — matrix operand order repaired and checked by reading, not compiled, not run on hardware
//
#include <hip/hip_runtime.h>
#include <hip/hip_bf16.h>
#include <math.h>

typedef _Float16 h16;
typedef __attribute__((ext_vector_type(16))) _Float16 v16h;
typedef __attribute__((ext_vector_type(8)))  _Float16 v8h;
typedef __attribute__((ext_vector_type(8)))  float    v8f;

#define B_   64
#define S_   1024
#define D_   768
#define T_   512
#define TP1  513
#define NPAD 528          // 513 padded to 33*16
#define NT2  33           // valid N tiles in GEMM2
#define MT   32           // 16-row M tiles (512/16), used by softmax/partials
#define KT   24           // K steps (768/32)
#define LDA  40           // LDS A-tile stride in halves (bank-conflict-free)
#define TSTRIDE (16 * D_) // halves between adjacent N tiles of B^T (24576 B imm)

__device__ __forceinline__ v16h cat8(v8h lo, v8h hi) {
    return __builtin_shufflevector(lo, hi, 0,1,2,3,4,5,6,7,8,9,10,11,12,13,14,15);
}

// A fragment (16x32 f16, MxK) from a 16xLDA LDS window:
// lanes 0-15: row M=lane, K 0..15; lanes 16-31: row M=lane-16, K 16..31
__device__ __forceinline__ v16h load_a_lds(const h16* As, int l, int hi) {
    const h16* p = As + l * LDA + hi * 16;
    v8h lo = *(const v8h*)p;
    v8h hh = *(const v8h*)(p + 8);
    return cat8(lo, hh);
}

// B fragment (32x16 f16, KxN) at a per-lane row pointer (row-major B^T):
// lanes 0-15: K=k..k+15 of column N=lane; lanes 16-31: K=k+16..k+31 (hi baked into p)
__device__ __forceinline__ v16h load_b_ptr(const h16* p) {
    __builtin_prefetch((const char*)p + 64, 0, 1);   // next K step -> global_prefetch_b8
    v8h lo = *(const v8h*)p;
    v8h hh = *(const v8h*)(p + 8);
    return cat8(lo, hh);
}

// Async direct-to-LDS staging (CDNA5 ASYNCcnt path). ldsb = low 32 bits of the
// flat shared address == hardware LDS byte address (ISA 10.2 aperture rule).
__device__ __forceinline__ void async_copy_b128(unsigned ldsb, const h16* src) {
    asm volatile("global_load_async_to_lds_b128 %0, %1, off"
                 :: "v"(ldsb), "v"(src) : "memory");
}

// --- 1) W[d][e] (f32) -> Wt[e][d] (f16) -------------------------------------
__global__ void wt_kernel(const float* __restrict__ W, h16* __restrict__ Wt) {
    int i = blockIdx.x * 256 + threadIdx.x;
    if (i < D_ * D_) {
        int e = i / D_, d = i - e * D_;
        Wt[i] = (h16)W[(size_t)d * D_ + e];
    }
}

// --- 2) ragged mean pool -> H f16 [B][NPAD][D]; row 0 = root, 513.. = 0 -----
__global__ void pool_kernel(const float* __restrict__ lh, const int* __restrict__ wid,
                            const float* __restrict__ root, h16* __restrict__ Hh,
                            int* __restrict__ nwords) {
    int b = blockIdx.y, tj = blockIdx.x, tid = threadIdx.x;
    h16* out = Hh + ((size_t)b * NPAD + tj) * D_;
    if (tj == 0) {
        for (int d = tid; d < D_; d += 256) out[d] = (h16)root[d];
        if (tid == 0) nwords[b] = wid[(size_t)b * S_ + (S_ - 1)];  // sorted -> max
        return;
    }
    if (tj > T_) {
        for (int d = tid; d < D_; d += 256) out[d] = (h16)0.0f;
        return;
    }
    int t = tj - 1;
    __shared__ int slo, shi;
    if (tid == 0) {
        const int* w = wid + (size_t)b * S_;
        int lo = 0, hc = S_;
        while (lo < hc) { int m = (lo + hc) >> 1; if (w[m] < t) lo = m + 1; else hc = m; }
        slo = lo;
        int lo2 = lo, h2 = S_;
        while (lo2 < h2) { int m = (lo2 + h2) >> 1; if (w[m] < t + 1) lo2 = m + 1; else h2 = m; }
        shi = lo2;
    }
    __syncthreads();
    int lo = slo, hiX = shi;
    float inv = (hiX > lo) ? 1.0f / (float)(hiX - lo) : 0.0f;   // empty segment -> 0
    const float* src = lh + (size_t)b * S_ * D_;
    for (int d = tid; d < D_; d += 256) {
        float s = 0.0f;
        for (int r = lo; r < hiX; ++r) s += src[(size_t)r * D_ + d];
        out[d] = (h16)(s * inv);
    }
}

// --- 3) u[b][j] = H[b][j] . U_w + U_b ---------------------------------------
__global__ void u_kernel(const h16* __restrict__ Hh, const float* __restrict__ Uw,
                         const float* __restrict__ Ub, float* __restrict__ u) {
    int j = blockIdx.x, b = blockIdx.y, lane = threadIdx.x;
    const h16* row = Hh + ((size_t)b * NPAD + j) * D_;
    float s = 0.0f;
    for (int k = lane; k < D_; k += 32) s += (float)row[k] * Uw[k];
    for (int m = 16; m; m >>= 1) s += __shfl_xor(s, m, 32);
    if (lane == 0) u[b * NPAD + j] = s + Ub[0];
}

// --- 4) XW[b] = X[b] @ W. Block: M=64 (4 subtiles), N=192 (2 grp x 6 tiles).
//     A tile double-buffered in LDS via global_load_async_to_lds_b128.
//     One per-lane B base pointer; tile offsets folded into 24-bit IOFFSET. --
__global__ void gemm1_kernel(const h16* __restrict__ Hh, const h16* __restrict__ Wt,
                             h16* __restrict__ XWh) {
    __shared__ __align__(16) h16 As[2][64 * LDA];
    int b = blockIdx.z, m0 = blockIdx.y * 64, ngrp = blockIdx.x;
    int tid = threadIdx.x, wave = tid >> 5, lane = tid & 31, l = lane & 15, hi = lane >> 4;
    int ms = wave >> 1, ng = wave & 1;
    const h16* Arows = Hh + ((size_t)b * NPAD + 1 + m0) * D_;   // X rows (skip root)
    int sr = tid >> 2, sc = (tid & 3) * 8;                      // one b128 per thread
    unsigned sdst[2] = { (unsigned)(size_t)&As[0][sr * LDA + sc],
                         (unsigned)(size_t)&As[1][sr * LDA + sc] };
    const h16* ssrc = Arows + (size_t)sr * D_ + sc;
    int nt0 = ngrp * 12 + ng * 6;
    const h16* bwave = Wt + (size_t)(nt0 * 16 + l) * D_ + hi * 16;  // tile 0, k=0
    v8f acc[6] = {};
    async_copy_b128(sdst[0], ssrc);                              // prologue
    for (int kt = 0; kt < KT; ++kt) {
        int cur = kt & 1;
        if (kt + 1 < KT) {
            async_copy_b128(sdst[cur ^ 1], ssrc + (kt + 1) * 32);
            asm volatile("s_wait_asynccnt 0x1" ::: "memory");    // current buf done
        } else {
            asm volatile("s_wait_asynccnt 0x0" ::: "memory");
        }
        __syncthreads();
        v16h a = load_a_lds(&As[cur][ms * 16 * LDA], l, hi);
#pragma unroll
        for (int t = 0; t < 6; ++t) {
            v16h bf = load_b_ptr(bwave + t * TSTRIDE);           // const imm offsets
            acc[t] = __builtin_amdgcn_wmma_f32_16x16x32_f16(false, a, false, bf,
                                                            (short)0, acc[t], false, false);
        }
        __syncthreads();
        bwave += 32;                                             // next K step
    }
    h16* Crow = XWh + ((size_t)b * T_ + m0 + ms * 16) * D_;
#pragma unroll
    for (int t = 0; t < 6; ++t) {
        int n = (nt0 + t) * 16 + l;
#pragma unroll
        for (int v = 0; v < 8; ++v)
            Crow[(size_t)(v + hi * 8) * D_ + n] = (h16)acc[t][v];
    }
}

// --- 5) bil[b] = XW[b] @ H[b]^T -> f32 global. Same blocking; 33 valid tiles.
__global__ void gemm2_kernel(const h16* __restrict__ XWh, const h16* __restrict__ Hh,
                             float* __restrict__ bil) {
    __shared__ __align__(16) h16 As[2][64 * LDA];
    int b = blockIdx.z, m0 = blockIdx.y * 64, ngrp = blockIdx.x;
    int tid = threadIdx.x, wave = tid >> 5, lane = tid & 31, l = lane & 15, hi = lane >> 4;
    int ms = wave >> 1, ng = wave & 1;
    const h16* Arows = XWh + ((size_t)b * T_ + m0) * D_;
    const h16* Brows = Hh + (size_t)b * NPAD * D_;
    int sr = tid >> 2, sc = (tid & 3) * 8;
    unsigned sdst[2] = { (unsigned)(size_t)&As[0][sr * LDA + sc],
                         (unsigned)(size_t)&As[1][sr * LDA + sc] };
    const h16* ssrc = Arows + (size_t)sr * D_ + sc;
    int nt0 = ngrp * 12 + ng * 6;
    const h16* bwave = Brows + (size_t)(nt0 * 16 + l) * D_ + hi * 16;
    v8f acc[6] = {};
    async_copy_b128(sdst[0], ssrc);
    for (int kt = 0; kt < KT; ++kt) {
        int cur = kt & 1;
        if (kt + 1 < KT) {
            async_copy_b128(sdst[cur ^ 1], ssrc + (kt + 1) * 32);
            asm volatile("s_wait_asynccnt 0x1" ::: "memory");
        } else {
            asm volatile("s_wait_asynccnt 0x0" ::: "memory");
        }
        __syncthreads();
        v16h a = load_a_lds(&As[cur][ms * 16 * LDA], l, hi);
#pragma unroll
        for (int t = 0; t < 6; ++t) {
            if (nt0 + t < NT2) {                                 // wave-uniform guard
                v16h bf = load_b_ptr(bwave + t * TSTRIDE);
                acc[t] = __builtin_amdgcn_wmma_f32_16x16x32_f16(false, a, false, bf,
                                                                (short)0, acc[t], false, false);
            }
        }
        __syncthreads();
        bwave += 32;
    }
    float* Crow = bil + ((size_t)b * T_ + m0 + ms * 16) * NPAD;
#pragma unroll
    for (int t = 0; t < 6; ++t) {
        int nt = nt0 + t;
        if (nt < NT2) {
#pragma unroll
            for (int v = 0; v < 8; ++v)
                Crow[(size_t)(v + hi * 8) * NPAD + nt * 16 + l] = acc[t][v];
        }
    }
}

// --- 6) masked log-softmax + NLL over bil(+u); deterministic partials -------
__global__ void softmax_kernel(const float* __restrict__ bil, const float* __restrict__ u,
                               const int* __restrict__ nwords, const int* __restrict__ Y,
                               float2* __restrict__ partials) {
    __shared__ float rowNll[16];
    __shared__ int   rowCnt[16];
    int b = blockIdx.y, mt = blockIdx.x, m0 = mt * 16;
    int tid = threadIdx.x, wave = tid >> 5, lane = tid & 31;
    int nw = nwords[b];
    const float* uu = u + b * NPAD;
    for (int rr = 0; rr < 2; ++rr) {
        int r = wave + rr * 8;             // 8 waves x 2 rows = 16 rows
        int t = m0 + r;
        float nll = 0.0f; int cnt = 0;
        if (t <= nw) {                      // valid dependent token
            const float* Lrow = bil + ((size_t)b * T_ + t) * NPAD;
            int jlim = nw + 1;              // head j valid iff j <= nw+1 (root j=0 incl.)
            float mx = -3.0e38f;
            for (int j = lane; j < TP1; j += 32) {
                float v = (j <= jlim) ? (Lrow[j] + uu[j]) : -1.0e9f;
                mx = fmaxf(mx, v);
            }
            for (int m = 16; m; m >>= 1) mx = fmaxf(mx, __shfl_xor(mx, m, 32));
            float se = 0.0f;
            for (int j = lane; j < TP1; j += 32) {
                float v = (j <= jlim) ? (Lrow[j] + uu[j]) : -1.0e9f;
                se += expf(v - mx);
            }
            for (int m = 16; m; m >>= 1) se += __shfl_xor(se, m, 32);
            float lse = mx + logf(se);
            int y = Y[(size_t)b * T_ + t];                      // 0..512
            float ly = (y <= jlim) ? (Lrow[y] + uu[y]) : -1.0e9f;
            nll = lse - ly; cnt = 1;
        }
        if (lane == 0) { rowNll[r] = nll; rowCnt[r] = cnt; }
    }
    __syncthreads();
    if (tid == 0) {
        float s = 0.0f; int c = 0;
        for (int r = 0; r < 16; ++r) { s += rowNll[r]; c += rowCnt[r]; }
        partials[b * MT + mt] = make_float2(s, (float)c);
    }
}

// --- 7) deterministic final reduction ---------------------------------------
__global__ void reduce_kernel(const float2* __restrict__ partials, float* __restrict__ out) {
    if (threadIdx.x == 0 && blockIdx.x == 0) {
        float num = 0.0f, den = 0.0f;
        for (int i = 0; i < B_ * MT; ++i) { num += partials[i].x; den += partials[i].y; }
        out[0] = num / fmaxf(den, 1.0f);
    }
}

extern "C" void kernel_launch(void* const* d_in, const int* in_sizes, int n_in,
                              void* d_out, int out_size, void* d_ws, size_t ws_size,
                              hipStream_t stream) {
    const float* lh   = (const float*)d_in[0];   // [B,S,D] f32
    const int*   wid  = (const int*)d_in[1];     // [B,S] int32 (JAX x64 off)
    const int*   Y    = (const int*)d_in[2];     // [B,T] int32
    const float* W    = (const float*)d_in[3];   // [D,D] f32
    const float* Uw   = (const float*)d_in[4];   // [D] f32
    const float* Ub   = (const float*)d_in[5];   // [1] f32
    const float* root = (const float*)d_in[6];   // [D] f32
    float* out = (float*)d_out;

    char* ws = (char*)d_ws;
    size_t off = 0;
    auto carve = [&](size_t bytes) -> void* {
        void* p = ws + off;
        off += (bytes + 255) & ~(size_t)255;
        return p;
    };
    h16*    Hh       = (h16*)   carve((size_t)B_ * NPAD * D_ * sizeof(h16));    // ~49.5 MB
    h16*    XWh      = (h16*)   carve((size_t)B_ * T_   * D_ * sizeof(h16));    // ~48.0 MB
    h16*    Wt       = (h16*)   carve((size_t)D_ * D_ * sizeof(h16));           // ~1.1 MB
    float*  bilbuf   = (float*) carve((size_t)B_ * T_ * NPAD * sizeof(float));  // ~66.0 MB
    float*  uvec     = (float*) carve((size_t)B_ * NPAD * sizeof(float));
    int*    nwords   = (int*)   carve((size_t)B_ * sizeof(int));
    float2* partials = (float2*)carve((size_t)B_ * MT * sizeof(float2));
    (void)in_sizes; (void)n_in; (void)out_size; (void)ws_size;

    wt_kernel     <<<(D_ * D_ + 255) / 256, 256, 0, stream>>>(W, Wt);
    pool_kernel   <<<dim3(NPAD, B_), 256, 0, stream>>>(lh, wid, root, Hh, nwords);
    u_kernel      <<<dim3(TP1, B_), 32, 0, stream>>>(Hh, Uw, Ub, uvec);
    gemm1_kernel  <<<dim3(4, 8, B_), 256, 0, stream>>>(Hh, Wt, XWh);
    gemm2_kernel  <<<dim3(3, 8, B_), 256, 0, stream>>>(XWh, Hh, bilbuf);
    softmax_kernel<<<dim3(MT, B_), 256, 0, stream>>>(bilbuf, uvec, nwords, Y, partials);
    reduce_kernel <<<1, 32, 0, stream>>>(partials, out);
}
